// MRM_56040733278680
// MI455X (gfx1250) — matrix operand order repaired and checked by reading, not compiled, not run on hardware
//
#include <hip/hip_runtime.h>
#include <hip/hip_bf16.h>
#include <math.h>

typedef __bf16 bf16;
typedef __attribute__((ext_vector_type(16))) __bf16 v16bf;
typedef __attribute__((ext_vector_type(8)))  float  v8f;
typedef int v4i __attribute__((vector_size(16)));
typedef __attribute__((address_space(1))) v4i* v4i_gptr;
typedef __attribute__((address_space(3))) v4i* v4i_lptr;

#define HW        81
#define NPIX      (HW*HW)          // 6561
#define BATCH     16
#define CIN       256
#define COUT      180
#define COUT_PAD  192
#define CLSX      20
#define KTOT      (CIN*9)          // 2304
#define KSTEPS    (KTOT/32)        // 72
#define MTILES    (COUT_PAD/16)    // 12
#define NTILE     48               // pixels per block (104976 / 48 = 2187 exactly)
#define NBLK      ((BATCH*NPIX)/NTILE)
#define CONV_THREADS 192           // 6 waves; each wave: 2 M-tiles x 3 N-frags
#define EPSB      1e-5f
#define ITERS     10

// gfx1250 async global->LDS path (builtin confirmed present by round-2 diagnostic)
#if defined(__has_builtin)
#if __has_builtin(__builtin_amdgcn_global_load_async_to_lds_b128) && \
    __has_builtin(__builtin_amdgcn_s_wait_asynccnt)
#define USE_ASYNC_LDS 1
#endif
#endif
#ifndef USE_ASYNC_LDS
#define USE_ASYNC_LDS 0
#endif

// ---------------- workspace layout (bytes) ----------------
static constexpr size_t SZ_ACT   = (size_t)BATCH*NPIX*CIN*2;            // bf16 NHWC activations
static constexpr size_t SZ_PACK  = (size_t)MTILES*KSTEPS*32*16*2;       // packed bf16 A fragments
static constexpr size_t SZ_SCSH  = 1024;                                 // 192 floats, padded
static constexpr size_t SZ_TLO   = (size_t)BATCH*41*41*256*4;           // low-res pointwise out (max)
static constexpr size_t OFF_ACT  = 0;
static constexpr size_t OFF_PACK = OFF_ACT  + SZ_ACT;
static constexpr size_t OFF_SC   = OFF_PACK + SZ_PACK;
static constexpr size_t OFF_SH   = OFF_SC   + SZ_SCSH;
static constexpr size_t OFF_TLO  = OFF_SH   + SZ_SCSH;
static constexpr size_t OFF_WACC = OFF_TLO  + SZ_TLO;

// ============================================================
// 1) NCHW f32 -> NHWC bf16 (branch 3 direct path)
// ============================================================
__global__ void nchw_to_nhwc_bf16(const float* __restrict__ src, bf16* __restrict__ act) {
  int total = BATCH*NPIX*CIN;
  for (int idx = blockIdx.x*blockDim.x + threadIdx.x; idx < total; idx += gridDim.x*blockDim.x) {
    int c    = idx & 255;
    int rest = idx >> 8;
    int pix  = rest % NPIX;
    int b    = rest / NPIX;
    act[idx] = (bf16)src[((size_t)b*CIN + c)*NPIX + pix];
  }
}

// ============================================================
// 2) pointwise 1x1 conv + bias + BN at LOW resolution (linear ops
//    commute with bilinear resize).  out layout: [b][p][o] f32
// ============================================================
__global__ void pw_bn(const float* __restrict__ fea, const float* __restrict__ lw,
                      const float* __restrict__ lb,  const float* __restrict__ bn,
                      float* __restrict__ tlo, int C, int P) {
  int total = BATCH*P*256;
  for (int idx = blockIdx.x*blockDim.x + threadIdx.x; idx < total; idx += gridDim.x*blockDim.x) {
    int o    = idx & 255;
    int rest = idx >> 8;
    int p    = rest % P;
    int b    = rest / P;
    const float* fr = fea + ((size_t)b*C)*P + p;
    const float* wr = lw + (size_t)o*C;
    float s = 0.f;
    for (int c = 0; c < C; ++c) s = fmaf(wr[c], fr[(size_t)c*P], s);
    float g = bn[o], bb = bn[256+o], m = bn[512+o], v = bn[768+o];
    float scale = g * rsqrtf(v + EPSB);
    tlo[idx] = scale * (s + lb[o] - m) + bb;
  }
}

// ============================================================
// 3) bilinear resize (align_corners style) [b][p][c] f32 -> NHWC bf16
// ============================================================
__global__ void resize_to_bf16(const float* __restrict__ tlo, bf16* __restrict__ act, int hl) {
  int total = BATCH*NPIX*CIN;
  float scale = (float)(hl - 1) / (float)(HW - 1);
  for (int idx = blockIdx.x*blockDim.x + threadIdx.x; idx < total; idx += gridDim.x*blockDim.x) {
    int c    = idx & 255;
    int rest = idx >> 8;
    int pix  = rest % NPIX;
    int b    = rest / NPIX;
    int y = pix / HW, x = pix % HW;
    float fy = y * scale, fx = x * scale;
    int y0 = (int)floorf(fy); int y1 = min(y0 + 1, hl - 1);
    int x0 = (int)floorf(fx); int x1 = min(x0 + 1, hl - 1);
    float wy = fy - (float)y0, wx = fx - (float)x0;
    const float* base = tlo + ((size_t)b*hl*hl)*CIN + c;
    float v00 = base[(size_t)(y0*hl + x0)*CIN];
    float v01 = base[(size_t)(y0*hl + x1)*CIN];
    float v10 = base[(size_t)(y1*hl + x0)*CIN];
    float v11 = base[(size_t)(y1*hl + x1)*CIN];
    float top = v00*(1.f-wx) + v01*wx;
    float bot = v10*(1.f-wx) + v11*wx;
    act[idx] = (bf16)(top*(1.f-wy) + bot*wy);
  }
}

// ============================================================
// 4) pack conv weights (OIHW f32) into per-lane WMMA A-fragment
//    layout (bf16 16x32 tiles, contiguous k = 16*(lane/16) + e), plus fold
//    bias+BN into per-channel scale/shift.  k = tap*256 + c.
// ============================================================
__global__ void pack_w(const float* __restrict__ w, const float* __restrict__ bias,
                       const float* __restrict__ bn, bf16* __restrict__ packA,
                       float* __restrict__ sc, float* __restrict__ sh) {
  int gid = blockIdx.x*blockDim.x + threadIdx.x;
  if (gid < COUT_PAD) {
    if (gid < COUT) {
      float g = bn[gid], bb = bn[COUT+gid], m = bn[2*COUT+gid], v = bn[3*COUT+gid];
      float s = g * rsqrtf(v + EPSB);
      sc[gid] = s;
      sh[gid] = bb + s * (bias[gid] - m);
    } else { sc[gid] = 0.f; sh[gid] = 0.f; }
  }
  int total = MTILES*KSTEPS*32*16;
  for (int idx = gid; idx < total; idx += gridDim.x*blockDim.x) {
    int e    = idx & 15;          // bf16 slot within lane's 8 VGPRs
    int lane = (idx >> 4) & 31;
    int rest = idx >> 9;
    int s    = rest % KSTEPS;
    int mt   = rest / KSTEPS;
    int j = e >> 1, sub = e & 1;
    int row, kk;
    if (lane < 16) { row = mt*16 + lane;        kk = 2*j; }
    else           { row = mt*16 + (lane - 16); kk = 16 + 2*j; }
    kk += sub;
    int kg  = s*32 + kk;
    int tap = kg >> 8;
    int c   = kg & 255;
    float val = (row < COUT) ? w[((size_t)row*CIN + c)*9 + tap] : 0.f;
    packA[idx] = (bf16)val;
  }
}

// ============================================================
// 5) implicit-GEMM 3x3 conv via WMMA bf16.
//    Block: 48 pixels x all 192 out-channels, 192 threads (6 waves).
//    Each wave: 2 M-tiles x 3 N-frags = 6 WMMA accumulators.
//    Double-buffered im2col tile in LDS; async global->LDS staging
//    (GLOBAL_LOAD_ASYNC_TO_LDS_B128 / ASYNCcnt), sync fallback otherwise.
// ============================================================
__device__ __forceinline__ void stage_tile(const bf16* __restrict__ act,
                                           unsigned short* dst, int n0, int s, int tid) {
  int tap = s >> 3;
  int dy = tap/3 - 1, dx = tap%3 - 1;
  int cbase = (s & 7) * 32;
  int p = tid >> 2, q = tid & 3;     // 48 pixels x 4 chunks of 16B == 192 threads
  int n = n0 + p;
  int b = n / NPIX, r = n % NPIX;
  int y = r / HW, x = r % HW;
  int yy = y + dy, xx = x + dx;
  bool valid = (yy >= 0 && yy < HW && xx >= 0 && xx < HW);
  const bf16* g = act + ((((size_t)b*NPIX + (size_t)yy*HW + xx) << 8) + cbase + q*8);
#if USE_ASYNC_LDS
  if (valid) {
    __builtin_amdgcn_global_load_async_to_lds_b128(
        (v4i_gptr)const_cast<bf16*>(g),
        (v4i_lptr)(dst + tid*8), 0, 0);
  } else {
    uint4 z = {0u,0u,0u,0u};
    *reinterpret_cast<uint4*>(dst + tid*8) = z;
  }
#else
  uint4 v = {0u,0u,0u,0u};
  if (valid) v = *reinterpret_cast<const uint4*>(g);
  *reinterpret_cast<uint4*>(dst + tid*8) = v;
#endif
}

__global__ __launch_bounds__(CONV_THREADS) void conv_wmma(
    const bf16* __restrict__ act, const bf16* __restrict__ packA,
    const float* __restrict__ sc, const float* __restrict__ sh,
    float* __restrict__ wacc, int initFlag) {
  __shared__ __align__(64) unsigned short btile[2][NTILE*32]; // 48 px x 32 bf16, double buffered
  __shared__ float ctile[COUT_PAD*NTILE];                     // 192 x 48 f32

  int tid  = threadIdx.x;
  int wave = tid >> 5;               // 0..5
  int lane = tid & 31;
  int n0   = blockIdx.x * NTILE;

  v8f acc[2][3];
#pragma unroll
  for (int i = 0; i < 2; ++i)
#pragma unroll
    for (int j = 0; j < 3; ++j) acc[i][j] = (v8f){0,0,0,0,0,0,0,0};

  const size_t mstride = (size_t)KSTEPS*32*16;              // bf16 elems per M-tile
  const bf16* abase = packA + (((size_t)(2*wave)*KSTEPS)*32 + lane)*16;

  // B-fragment LDS offsets (ushort units): frag j, pixel = j*16 + lane%16, k-half = lane/16
  int boff0 = ((0*16) + (lane & 15))*32 + (lane >> 4)*16;
  int boff1 = ((1*16) + (lane & 15))*32 + (lane >> 4)*16;
  int boff2 = ((2*16) + (lane & 15))*32 + (lane >> 4)*16;

  // prologue: stage tile 0
  stage_tile(act, btile[0], n0, 0, tid);
#if USE_ASYNC_LDS
  __builtin_amdgcn_s_wait_asynccnt(0);
#endif
  __syncthreads();

  for (int s = 0; s < KSTEPS; ++s) {
    unsigned short* cur = btile[s & 1];
    if (s + 1 < KSTEPS) stage_tile(act, btile[(s + 1) & 1], n0, s + 1, tid);

    v16bf b0 = *reinterpret_cast<const v16bf*>(&cur[boff0]);
    v16bf b1 = *reinterpret_cast<const v16bf*>(&cur[boff1]);
    v16bf b2 = *reinterpret_cast<const v16bf*>(&cur[boff2]);

    const bf16* ap = abase + (size_t)s*32*16;
    if (s + 1 < KSTEPS) __builtin_prefetch(ap + 32*16, 0, 1);  // global_prefetch_b8
    v16bf a0 = *reinterpret_cast<const v16bf*>(ap);
    v16bf a1 = *reinterpret_cast<const v16bf*>(ap + mstride);

    acc[0][0] = __builtin_amdgcn_wmma_f32_16x16x32_bf16(false, a0, false, b0, (short)0, acc[0][0], false, false);
    acc[0][1] = __builtin_amdgcn_wmma_f32_16x16x32_bf16(false, a0, false, b1, (short)0, acc[0][1], false, false);
    acc[0][2] = __builtin_amdgcn_wmma_f32_16x16x32_bf16(false, a0, false, b2, (short)0, acc[0][2], false, false);
    acc[1][0] = __builtin_amdgcn_wmma_f32_16x16x32_bf16(false, a1, false, b0, (short)0, acc[1][0], false, false);
    acc[1][1] = __builtin_amdgcn_wmma_f32_16x16x32_bf16(false, a1, false, b1, (short)0, acc[1][1], false, false);
    acc[1][2] = __builtin_amdgcn_wmma_f32_16x16x32_bf16(false, a1, false, b2, (short)0, acc[1][2], false, false);

#if USE_ASYNC_LDS
    __builtin_amdgcn_s_wait_asynccnt(0);
#endif
    __syncthreads();
  }

  // scatter accumulators to LDS per ISA C-layout: VGPR r -> M = r + 8*(lane>=16), N = lane%16
  int col = lane & 15;
  int rb  = (lane >> 4) * 8;
#pragma unroll
  for (int i = 0; i < 2; ++i) {
    int rowb = (2*wave + i)*16 + rb;
#pragma unroll
    for (int j = 0; j < 3; ++j) {
      int cb = j*16 + col;
#pragma unroll
      for (int r = 0; r < 8; ++r)
        ctile[(rowb + r)*NTILE + cb] = acc[i][j][r];
    }
  }
  __syncthreads();

  // BN + ReLU + per-class tap normalization, accumulate (w/sum)/3
  for (int t = tid; t < CLSX*NTILE; t += CONV_THREADS) {
    int cls = t / NTILE, px = t % NTILE;
    int n = n0 + px;
    int b = n / NPIX, r = n % NPIX;
    float vals[9]; float ssum = 0.f;
#pragma unroll
    for (int tap = 0; tap < 9; ++tap) {
      int o = cls*9 + tap;
      float v = sc[o]*ctile[o*NTILE + px] + sh[o];
      v = v > 0.f ? v : 0.f;
      vals[tap] = v; ssum += v;
    }
    if (ssum == 0.f) ssum = 1.f;
    float inv = (1.f/3.f) / ssum;
    size_t base = (((size_t)b*CLSX + cls)*9)*NPIX + r;
#pragma unroll
    for (int tap = 0; tap < 9; ++tap) {
      size_t o = base + (size_t)tap*NPIX;
      float w = vals[tap]*inv;
      if (initFlag) wacc[o] = w; else wacc[o] += w;
    }
  }
}

// ============================================================
// 6) 10 filtering iterations, one (b,class) plane per block.
//    9 weight planes + double-buffered x plane in LDS (~282 KB,
//    exploits CDNA5's 320KB/WGP LDS).
// ============================================================
__global__ void iterate_kernel(const float* __restrict__ x0,
                               const float* __restrict__ weight,
                               float* __restrict__ out) {
  extern __shared__ float smem[];
  float* wsm = smem;               // 9*NPIX
  float* xa  = smem + 9*NPIX;      // NPIX
  float* xb  = xa + NPIX;          // NPIX
  int plane = blockIdx.x;
  size_t wbase = (size_t)plane * 9 * NPIX;
  size_t xbase = (size_t)plane * NPIX;
  for (int i = threadIdx.x; i < 9*NPIX; i += blockDim.x) wsm[i] = weight[wbase + i];
  for (int i = threadIdx.x; i < NPIX;  i += blockDim.x) xa[i] = x0[xbase + i];
  __syncthreads();
  float* cur = xa; float* nxt = xb;
  for (int it = 0; it < ITERS; ++it) {
    for (int i = threadIdx.x; i < NPIX; i += blockDim.x) {
      int y = i / HW, x = i % HW;
      float s = 0.f;
#pragma unroll
      for (int tap = 0; tap < 9; ++tap) {
        int yy = y + tap/3 - 1, xx = x + tap%3 - 1;
        float xv = (yy >= 0 && yy < HW && xx >= 0 && xx < HW) ? cur[yy*HW + xx] : 0.f;
        s = fmaf(xv, wsm[(size_t)tap*NPIX + i], s);
      }
      nxt[i] = s;
    }
    __syncthreads();
    float* tmp = cur; cur = nxt; nxt = tmp;
  }
  for (int i = threadIdx.x; i < NPIX; i += blockDim.x) out[xbase + i] = cur[i];
}

// ============================================================
// launcher
// ============================================================
extern "C" void kernel_launch(void* const* d_in, const int* in_sizes, int n_in,
                              void* d_out, int out_size, void* d_ws, size_t ws_size,
                              hipStream_t stream) {
  (void)in_sizes; (void)n_in; (void)out_size; (void)ws_size;
  const float* x     = (const float*)d_in[0];
  const float* fea0  = (const float*)d_in[1];
  const float* fea1  = (const float*)d_in[2];
  const float* fea2  = (const float*)d_in[3];
  const float* l1w   = (const float*)d_in[4];
  const float* l1b   = (const float*)d_in[5];
  const float* bnl1  = (const float*)d_in[6];
  const float* l2w   = (const float*)d_in[7];
  const float* l2b   = (const float*)d_in[8];
  const float* bnl2  = (const float*)d_in[9];
  const float* s1w   = (const float*)d_in[10];
  const float* s1b   = (const float*)d_in[11];
  const float* bns1  = (const float*)d_in[12];
  const float* s2w   = (const float*)d_in[13];
  const float* s2b   = (const float*)d_in[14];
  const float* bns2  = (const float*)d_in[15];
  const float* s3w   = (const float*)d_in[16];
  const float* s3b   = (const float*)d_in[17];
  const float* bns3  = (const float*)d_in[18];

  char* ws = (char*)d_ws;
  bf16*  act   = (bf16*)(ws + OFF_ACT);
  bf16*  packA = (bf16*)(ws + OFF_PACK);
  float* scb   = (float*)(ws + OFF_SC);
  float* shb   = (float*)(ws + OFF_SH);
  float* tlo   = (float*)(ws + OFF_TLO);
  float* wacc  = (float*)(ws + OFF_WACC);
  float* outp  = (float*)d_out;

  const int actTotal = BATCH*NPIX*CIN;
  const int actBlocks = (actTotal + 255)/256;

  // ---- branch 3 (direct): conv(fea0) ----
  nchw_to_nhwc_bf16<<<actBlocks, 256, 0, stream>>>(fea0, act);
  pack_w<<<1728, 256, 0, stream>>>(s1w, s1b, bns1, packA, scb, shb);
  conv_wmma<<<NBLK, CONV_THREADS, 0, stream>>>(act, packA, scb, shb, wacc, 1);

  // ---- branch 2: fea1 (512ch @41x41) ----
  {
    int P = 41*41;
    int total = BATCH*P*256;
    pw_bn<<<(total + 255)/256, 256, 0, stream>>>(fea1, l2w, l2b, bnl2, tlo, 512, P);
    resize_to_bf16<<<actBlocks, 256, 0, stream>>>(tlo, act, 41);
    pack_w<<<1728, 256, 0, stream>>>(s2w, s2b, bns2, packA, scb, shb);
    conv_wmma<<<NBLK, CONV_THREADS, 0, stream>>>(act, packA, scb, shb, wacc, 0);
  }

  // ---- branch 1: fea2 (1024ch @21x21) ----
  {
    int P = 21*21;
    int total = BATCH*P*256;
    pw_bn<<<(total + 255)/256, 256, 0, stream>>>(fea2, l1w, l1b, bnl1, tlo, 1024, P);
    resize_to_bf16<<<actBlocks, 256, 0, stream>>>(tlo, act, 21);
    pack_w<<<1728, 256, 0, stream>>>(s3w, s3b, bns3, packA, scb, shb);
    conv_wmma<<<NBLK, CONV_THREADS, 0, stream>>>(act, packA, scb, shb, wacc, 0);
  }

  // ---- iterative per-pixel filtering ----
  const int smemBytes = (9 + 2) * NPIX * (int)sizeof(float); // 288,684 B
  (void)hipFuncSetAttribute(reinterpret_cast<const void*>(iterate_kernel),
                            hipFuncAttributeMaxDynamicSharedMemorySize, smemBytes);
  iterate_kernel<<<BATCH*CLSX, 256, smemBytes, stream>>>(x, wacc, outp);
}
